// GCN_5403068859072
// MI455X (gfx1250) — matrix operand order repaired and checked by reading, not compiled, not run on hardware
//
#include <hip/hip_runtime.h>

// ---------------------------------------------------------------------------
// CDNA5 (gfx1250, wave32) fused GCN.
//
// bf16 WMMA (v_wmma_f32_16x16x32_bf16), f32 accumulate.
// Message bias folded into apply stage:  h = (sum A@W)*inv_deg + b*(deg>0).
// Message tensor (E x D) never materialized: WMMA tile -> atomicAdd scatter.
// Scatter uses LDS-staged 32-bit element indices against the uniform accum
// base so atomics take the SADDR+VOFFSET (scale_offset) addressing form.
// ---------------------------------------------------------------------------

typedef __bf16 bf16_t;
typedef bf16_t v16bf __attribute__((ext_vector_type(16)));
typedef float  v8f   __attribute__((ext_vector_type(8)));

union FragU { v16bf v; uint4 u[2]; };

// A-fragment (16x32 bf16): lane<16 -> row M=lane, K {k..k+15};
// lanes>=16 hold K {k+16..k+31} (base +16).
__device__ inline v16bf lds_frag_A(const bf16_t* rowPtr, int kbase) {
  FragU f;
  f.u[0] = *reinterpret_cast<const uint4*>(rowPtr + kbase);
  f.u[1] = *reinterpret_cast<const uint4*>(rowPtr + kbase + 8);
  return f.v;
}

// B-fragment: pre-swizzled in global memory, 32 contiguous bytes per lane.
__device__ inline v16bf glb_frag_B(const bf16_t* p) {
  FragU f;
  const uint4* q = reinterpret_cast<const uint4*>(p);
  f.u[0] = q[0];
  f.u[1] = q[1];
  return f.v;
}

__device__ inline uint4 pack8_bf16(const float* v) {
  union { bf16_t h[8]; uint4 u; } P;
#pragma unroll
  for (int j = 0; j < 8; ++j) P.h[j] = (bf16_t)v[j];
  return P.u;
}

// ---------------------------------------------------------------------------
// Weight pre-swizzle: W[K][DOUT] f32 row-major -> bf16 B-fragment stream.
// Block (kt,nt) = 512 elems; elem (lane,s): K = kt*32 + 16*(lane>>4) + s,
// N = nt*16 + (lane&15).  (32x16 bf16 B layout, wave32.)
// ---------------------------------------------------------------------------
__global__ __launch_bounds__(256) void swizzle_w(const float* __restrict__ W,
                                                 bf16_t* __restrict__ out,
                                                 int K, int DOUT) {
  int idx = blockIdx.x * blockDim.x + threadIdx.x;
  if (idx >= K * DOUT) return;
  int NT   = DOUT >> 4;
  int blk  = idx >> 9;
  int r    = idx & 511;
  int lane = r >> 4;
  int s    = r & 15;
  int kt   = blk / NT;
  int nt   = blk - kt * NT;
  int k    = kt * 32 + 16 * (lane >> 4) + s;
  int n    = nt * 16 + (lane & 15);
  out[idx] = (bf16_t)W[(size_t)k * DOUT + n];
}

__global__ __launch_bounds__(256) void zero_kernel(float* __restrict__ p, size_t n) {
  size_t i = (size_t)blockIdx.x * blockDim.x + threadIdx.x;
  if (i < n) p[i] = 0.f;
}

__global__ __launch_bounds__(256) void degree_kernel(const int* __restrict__ dst,
                                                     float* __restrict__ deg,
                                                     int n_edges) {
  int e = blockIdx.x * blockDim.x + threadIdx.x;
  if (e < n_edges) atomicAdd(&deg[dst[e]], 1.0f);
}

// ---------------------------------------------------------------------------
// Fused message GEMM + scatter.
//   A (16 edges x K) = [ x[src] | e ],  K = DX + 64
//   per-wave: one 16-col output tile; KT = K/32 wmma steps; atomicAdd scatter.
// Block = 256 threads = 8 waves = EG edge-groups x NT n-tiles.
// ---------------------------------------------------------------------------
template <int DX, int NT, int EG>
__global__ __launch_bounds__(256) void msg_kernel(
    const float* __restrict__ xfeat, const float* __restrict__ efeat,
    const int* __restrict__ src, const int* __restrict__ dst,
    const bf16_t* __restrict__ Wswz, float* __restrict__ accum, int n_edges) {
  constexpr int K   = DX + 64;
  constexpr int KT  = K / 32;
  constexpr int DO  = NT * 16;
  constexpr int PAD = 8;  // 16B row pad -> breaks 64-bank stride conflicts
  __shared__ bf16_t Atile[EG][16][K + PAD];
  __shared__ int    dstOff[EG * 16];  // dst[edge] * DO, staged once

  const int tid  = threadIdx.x;
  const int lane = tid & 31;
  const int wave = tid >> 5;
  const int eg   = wave / NT;
  const int nt   = wave % NT;
  const int blk_edge0 = blockIdx.x * (16 * EG);
  const bool tile_full = (blk_edge0 + 16 * EG) <= n_edges;  // uniform

  if (tid < EG * 16) {
    const int e = blk_edge0 + tid;
    dstOff[tid] = (e < n_edges) ? dst[e] * DO : -(1 << 28);
  }

  // Cooperative gather: fp32 -> bf16, 8-elem (16B LDS store) chunks.
  constexpr int CH = K / 8;
  auto fill = [&](bool guard) {
    for (int idx = tid; idx < EG * 16 * CH; idx += 256) {
      const int g    = idx / (16 * CH);
      const int rem  = idx - g * (16 * CH);
      const int r    = rem / CH;
      const int c8   = (rem - r * CH) * 8;
      const int edge = blk_edge0 + g * 16 + r;
      float v[8];
      if (!guard || edge < n_edges) {
        const float* p = (c8 < DX) ? xfeat + (size_t)src[edge] * DX + c8
                                   : efeat + (size_t)edge * 64 + (c8 - DX);
        float4 a = *reinterpret_cast<const float4*>(p);
        float4 b = *reinterpret_cast<const float4*>(p + 4);
        v[0] = a.x; v[1] = a.y; v[2] = a.z; v[3] = a.w;
        v[4] = b.x; v[5] = b.y; v[6] = b.z; v[7] = b.w;
      } else {
#pragma unroll
        for (int j = 0; j < 8; ++j) v[j] = 0.f;
      }
      *reinterpret_cast<uint4*>(&Atile[g][r][c8]) = pack8_bf16(v);
    }
  };
  if (tile_full) fill(false); else fill(true);
  __syncthreads();

  // WMMA: (16 x K) @ (K x 16)
  v8f acc = {};
  const bf16_t* arow = &Atile[eg][lane & 15][0];
  const int k0 = 16 * (lane >> 4);
#pragma unroll
  for (int kt = 0; kt < KT; ++kt) {
    v16bf a = lds_frag_A(arow, kt * 32 + k0);
    v16bf b = glb_frag_B(Wswz + (size_t)(kt * NT + nt) * 512 + lane * 16);
    acc = __builtin_amdgcn_wmma_f32_16x16x32_bf16(false, a, false, b,
                                                  (short)0, acc, false, false);
  }

  // Scatter-add. D layout: vgpr i, lanes<16 -> M=i, lanes>=16 -> M=i+8;
  // N = lane&15.  Two ds_load_b128 grab all 8 staged indices; atomics index
  // the uniform `accum` base with 32-bit offsets (SADDR+VOFFSET form).
  const int col  = nt * 16 + (lane & 15);
  const int row0 = eg * 16 + 8 * (lane >> 4);   // multiple of 8 -> 32B aligned
  const int4* dp = reinterpret_cast<const int4*>(&dstOff[row0]);
  const int4 o0 = dp[0];
  const int4 o1 = dp[1];
  int ofs[8] = {o0.x, o0.y, o0.z, o0.w, o1.x, o1.y, o1.z, o1.w};
  if (tile_full) {
#pragma unroll
    for (int i = 0; i < 8; ++i)
      atomicAdd(&accum[(unsigned)(ofs[i] + col)], acc[i]);
  } else {
#pragma unroll
    for (int i = 0; i < 8; ++i)
      if (ofs[i] >= 0) atomicAdd(&accum[(unsigned)(ofs[i] + col)], acc[i]);
  }
}

// ---------------------------------------------------------------------------
// Apply stage: out = relu([x | accum*inv_deg + b_msg*(deg>0)] @ Wa + b_apply)
// ---------------------------------------------------------------------------
template <int DX, int DH, int NT, int NG>
__global__ __launch_bounds__(256) void apply_kernel(
    const float* __restrict__ xfeat, const float* __restrict__ accum,
    const float* __restrict__ deg, const float* __restrict__ bmsg,
    const bf16_t* __restrict__ Wswz, const float* __restrict__ bapply,
    float* __restrict__ out, int n_nodes) {
  constexpr int K   = DX + DH;
  constexpr int KT  = K / 32;
  constexpr int DO  = NT * 16;
  constexpr int PAD = 8;
  __shared__ bf16_t Atile[NG][16][K + PAD];

  const int tid  = threadIdx.x;
  const int lane = tid & 31;
  const int wave = tid >> 5;
  const int ng   = wave / NT;
  const int nt   = wave % NT;
  const int blk_node0 = blockIdx.x * (16 * NG);
  const bool tile_full = (blk_node0 + 16 * NG) <= n_nodes;  // uniform

  constexpr int CH = K / 8;
  auto fill = [&](bool guard) {
    for (int idx = tid; idx < NG * 16 * CH; idx += 256) {
      const int g    = idx / (16 * CH);
      const int rem  = idx - g * (16 * CH);
      const int r    = rem / CH;
      const int c8   = (rem - r * CH) * 8;
      const int node = blk_node0 + g * 16 + r;
      float v[8];
      if (!guard || node < n_nodes) {
        if (c8 < DX) {
          const float* p = xfeat + (size_t)node * DX + c8;
          float4 a = *reinterpret_cast<const float4*>(p);
          float4 b = *reinterpret_cast<const float4*>(p + 4);
          v[0] = a.x; v[1] = a.y; v[2] = a.z; v[3] = a.w;
          v[4] = b.x; v[5] = b.y; v[6] = b.z; v[7] = b.w;
        } else {
          const int hc   = c8 - DX;
          const float d  = deg[node];
          const float iv = d > 0.f ? 1.f / d : 0.f;
          const float bo = d > 0.f ? 1.f : 0.f;
          const float* p = accum + (size_t)node * DH + hc;
#pragma unroll
          for (int j = 0; j < 8; ++j) v[j] = p[j] * iv + bmsg[hc + j] * bo;
        }
      } else {
#pragma unroll
        for (int j = 0; j < 8; ++j) v[j] = 0.f;
      }
      *reinterpret_cast<uint4*>(&Atile[g][r][c8]) = pack8_bf16(v);
    }
  };
  if (tile_full) fill(false); else fill(true);
  __syncthreads();

  v8f acc = {};
  const bf16_t* arow = &Atile[ng][lane & 15][0];
  const int k0 = 16 * (lane >> 4);
#pragma unroll
  for (int kt = 0; kt < KT; ++kt) {
    v16bf a = lds_frag_A(arow, kt * 32 + k0);
    v16bf b = glb_frag_B(Wswz + (size_t)(kt * NT + nt) * 512 + lane * 16);
    acc = __builtin_amdgcn_wmma_f32_16x16x32_bf16(false, a, false, b,
                                                  (short)0, acc, false, false);
  }

  const int   col = nt * 16 + (lane & 15);
  const int   m0  = ng * 16 + 8 * (lane >> 4);
  const float bc  = bapply[col];
  const unsigned obase = (unsigned)(blk_node0 + m0) * DO + col;
  if (tile_full) {
#pragma unroll
    for (int i = 0; i < 8; ++i) {
      float r = acc[i] + bc;
      out[obase + (unsigned)i * DO] = r > 0.f ? r : 0.f;
    }
  } else {
#pragma unroll
    for (int i = 0; i < 8; ++i) {
      if (blk_node0 + m0 + i < n_nodes) {
        float r = acc[i] + bc;
        out[obase + (unsigned)i * DO] = r > 0.f ? r : 0.f;
      }
    }
  }
}

// ---------------------------------------------------------------------------
extern "C" void kernel_launch(void* const* d_in, const int* in_sizes, int n_in,
                              void* d_out, int out_size, void* d_ws, size_t ws_size,
                              hipStream_t stream) {
  const float* nfeats = (const float*)d_in[0];
  const float* efeats = (const float*)d_in[1];
  const int*   src    = (const int*)d_in[2];
  const int*   dst    = (const int*)d_in[3];
  const float* W1m    = (const float*)d_in[4];
  const float* b1m    = (const float*)d_in[5];
  const float* W1a    = (const float*)d_in[6];
  const float* b1a    = (const float*)d_in[7];
  const float* W2m    = (const float*)d_in[8];
  const float* b2m    = (const float*)d_in[9];
  const float* W2a    = (const float*)d_in[10];
  const float* b2a    = (const float*)d_in[11];

  const int N = in_sizes[0] / 64;
  const int E = in_sizes[2];

  auto al = [](size_t x) { return (x + 255) & ~(size_t)255; };
  char* ws = (char*)d_ws;
  size_t off = 0;
  float*  deg = (float*)(ws + off);  off = al(off + (size_t)N * 4);
  float*  h1  = (float*)(ws + off);  off = al(off + (size_t)N * 128 * 4);
  float*  h2  = (float*)(ws + off);  off = al(off + (size_t)N * 64 * 4);
  float*  x1  = (float*)(ws + off);  off = al(off + (size_t)N * 128 * 4);
  bf16_t* w1m = (bf16_t*)(ws + off); off = al(off + (size_t)128 * 128 * 2);
  bf16_t* w1a = (bf16_t*)(ws + off); off = al(off + (size_t)192 * 128 * 2);
  bf16_t* w2m = (bf16_t*)(ws + off); off = al(off + (size_t)192 * 64 * 2);
  bf16_t* w2a = (bf16_t*)(ws + off); off = al(off + (size_t)192 * 64 * 2);

  auto blocks = [](size_t n, int b) { return (unsigned)((n + b - 1) / b); };

  // Zero accumulators (+degree).
  zero_kernel<<<blocks((size_t)N, 256), 256, 0, stream>>>(deg, (size_t)N);
  zero_kernel<<<blocks((size_t)N * 128, 256), 256, 0, stream>>>(h1, (size_t)N * 128);
  zero_kernel<<<blocks((size_t)N * 64, 256), 256, 0, stream>>>(h2, (size_t)N * 64);

  // Pre-swizzle weights -> bf16 WMMA B fragments.
  swizzle_w<<<blocks(128 * 128, 256), 256, 0, stream>>>(W1m, w1m, 128, 128);
  swizzle_w<<<blocks(192 * 128, 256), 256, 0, stream>>>(W1a, w1a, 192, 128);
  swizzle_w<<<blocks(192 * 64, 256), 256, 0, stream>>>(W2m, w2m, 192, 64);
  swizzle_w<<<blocks(192 * 64, 256), 256, 0, stream>>>(W2a, w2a, 192, 64);

  // Degrees.
  degree_kernel<<<blocks((size_t)E, 256), 256, 0, stream>>>(dst, deg, E);

  // Layer 1: K=128, DO=128 -> NT=8, EG=1 (16 edges/WG).
  msg_kernel<64, 8, 1><<<blocks((size_t)E, 16), 256, 0, stream>>>(
      nfeats, efeats, src, dst, w1m, h1, E);
  apply_kernel<64, 128, 8, 1><<<blocks((size_t)N, 16), 256, 0, stream>>>(
      nfeats, h1, deg, b1m, w1a, b1a, x1, N);

  // Layer 2: K=192, DO=64 -> NT=4, EG=2 (32 edges/WG).
  msg_kernel<128, 4, 2><<<blocks((size_t)E, 32), 256, 0, stream>>>(
      x1, efeats, src, dst, w2m, h2, E);
  apply_kernel<128, 64, 4, 2><<<blocks((size_t)N, 32), 256, 0, stream>>>(
      x1, h2, deg, b2m, w2a, b2a, (float*)d_out, N);
}
